// QwenStyleSparseMoEBlock_RL_83597243449392
// MI455X (gfx1250) — compile-verified, matrix operand order checked by reading, not run on hardware
//
#include <hip/hip_runtime.h>
#include <hip/hip_bf16.h>
#include <math.h>
#include <stdint.h>

#define DMODEL 1024
#define DFF    2048
#define NE     8
#define TOPK   2

typedef __attribute__((ext_vector_type(16))) __bf16 v16bf;
typedef __attribute__((ext_vector_type(8)))  float  v8f;

union FragBF16 { v16bf v; uint4 u[2]; };
union Pack8BF  { __bf16 h[8]; uint4 u; };

// LDS byte offset of a __shared__ pointer (LDS aperture: addr[31:0] = offset)
__device__ __forceinline__ uint32_t ldsU32(const void* p) {
    return (uint32_t)(uintptr_t)p;
}

// CDNA5 async copy: 16B per lane, global -> LDS, tracked by ASYNCcnt.
__device__ __forceinline__ void asyncCp16(uint32_t lds_addr, const void* gptr) {
    asm volatile("global_load_async_to_lds_b128 %0, %1, off"
                 :: "v"(lds_addr), "v"(gptr) : "memory");
}
__device__ __forceinline__ void waitAsync0() {
    asm volatile("s_wait_asynccnt 0x0" ::: "memory");
}

// Load one 16x32 bf16 WMMA fragment from LDS tile row-major [rows][32] bf16.
// Lanes 0-15: K=0..7,16..23; lanes 16-31: K=8..15,24..31 -> 2x ds_load_b128.
__device__ __forceinline__ v16bf ldsFrag(const __bf16* tile, int rbase, int lane) {
    const int half = lane >> 4;
    const int r    = rbase + (lane & 15);
    const char* p  = (const char*)(tile + r * 32) + half * 16;
    FragBF16 f;
    f.u[0] = *(const uint4*)(p);
    f.u[1] = *(const uint4*)(p + 32);
    return f.v;
}

// ---------------------------------------------------------------- router ----
__global__ __launch_bounds__(256)
void moe_router(const float* __restrict__ x, const float* __restrict__ gw,
                int* __restrict__ counts, int* __restrict__ idx,
                float* __restrict__ wgt, int N) {
    const int wave = threadIdx.x >> 5;
    const int lane = threadIdx.x & 31;
    const int token = blockIdx.x * 8 + wave;
    if (token >= N) return;

    float acc[NE];
#pragma unroll
    for (int e = 0; e < NE; ++e) acc[e] = 0.f;

    const float* xr = x + (size_t)token * DMODEL;
    for (int k = lane; k < DMODEL; k += 32) {
        const float xv = xr[k];
        const float* g = gw + k * NE;
#pragma unroll
        for (int e = 0; e < NE; ++e) acc[e] += xv * g[e];
    }
#pragma unroll
    for (int e = 0; e < NE; ++e) {
#pragma unroll
        for (int off = 16; off > 0; off >>= 1)
            acc[e] += __shfl_down(acc[e], off, 32);
    }
    if (lane == 0) {
        float mx = acc[0];
#pragma unroll
        for (int e = 1; e < NE; ++e) mx = fmaxf(mx, acc[e]);
        float p[NE], s = 0.f;
#pragma unroll
        for (int e = 0; e < NE; ++e) { p[e] = __expf(acc[e] - mx); s += p[e]; }
        const float inv = 1.f / s;
#pragma unroll
        for (int e = 0; e < NE; ++e) p[e] *= inv;
        int i0 = 0;
#pragma unroll
        for (int e = 1; e < NE; ++e) if (p[e] > p[i0]) i0 = e;
        int i1 = (i0 == 0) ? 1 : 0;
#pragma unroll
        for (int e = 0; e < NE; ++e) if (e != i0 && p[e] > p[i1]) i1 = e;
        const float sn = p[i0] + p[i1];
        const float w0 = p[i0] / sn, w1 = p[i1] / sn;
        int pos = atomicAdd(&counts[i0], 1);
        idx[i0 * N + pos] = token;  wgt[i0 * N + pos] = w0;
        pos = atomicAdd(&counts[i1], 1);
        idx[i1 * N + pos] = token;  wgt[i1 * N + pos] = w1;
    }
}

__global__ void moe_prefix(const int* __restrict__ counts, int* __restrict__ offs) {
    if (threadIdx.x == 0) {
        int a = 0;
#pragma unroll
        for (int e = 0; e < NE; ++e) { offs[e] = a; a += counts[e]; }
    }
}

// ------------------------------------------------------- one-time repack ----
// X: fp32 -> bf16, linear.
__global__ __launch_bounds__(256)
void pack_x(const float* __restrict__ in, __bf16* __restrict__ out) {
    const size_t base = ((size_t)blockIdx.x * 256 + threadIdx.x) * 8;
    const float4 a = *(const float4*)(in + base);
    const float4 b = *(const float4*)(in + base + 4);
    Pack8BF o;
    o.h[0] = (__bf16)a.x; o.h[1] = (__bf16)a.y; o.h[2] = (__bf16)a.z; o.h[3] = (__bf16)a.w;
    o.h[4] = (__bf16)b.x; o.h[5] = (__bf16)b.y; o.h[6] = (__bf16)b.z; o.h[7] = (__bf16)b.w;
    *(uint4*)(out + base) = o.u;
}

// Per-expert [K][Nc] fp32 -> [Nc][K] bf16 (LDS-tiled transpose, 64x64 tiles).
__global__ __launch_bounds__(256)
void pack_transpose(const float* __restrict__ in, __bf16* __restrict__ out,
                    int K, int Nc) {
    const size_t eoff = (size_t)blockIdx.z * K * Nc;
    const float* ip  = in  + eoff;
    __bf16*      op  = out + eoff;
    const int kb = blockIdx.x * 64;
    const int nb = blockIdx.y * 64;

    __shared__ __bf16 T[64][72];
    const int tid = threadIdx.x;
#pragma unroll
    for (int j = 0; j < 4; ++j) {
        const int lin = tid * 4 + j;            // 0..1023 float4s
        const int r = lin >> 4, c4 = lin & 15;
        const float4 v = *(const float4*)(ip + (size_t)(kb + r) * Nc + nb + c4 * 4);
        T[r][c4 * 4 + 0] = (__bf16)v.x;
        T[r][c4 * 4 + 1] = (__bf16)v.y;
        T[r][c4 * 4 + 2] = (__bf16)v.z;
        T[r][c4 * 4 + 3] = (__bf16)v.w;
    }
    __syncthreads();
#pragma unroll
    for (int j = 0; j < 2; ++j) {
        const int lin = tid * 2 + j;            // 0..511 16B chunks
        const int n = lin >> 3, c8 = lin & 7;
        Pack8BF o;
#pragma unroll
        for (int q = 0; q < 8; ++q) o.h[q] = T[c8 * 8 + q][n];
        *(uint4*)(op + (size_t)(nb + n) * K + kb + c8 * 8) = o.u;
    }
}

// -------------------------------------------------- GEMM1: gate+up+GELU ----
// All operands bf16 panels; staging = async global->LDS b128; double-buffered.
__global__ __launch_bounds__(256)
void moe_gemm_gateup(const __bf16* __restrict__ Xb,
                     const __bf16* __restrict__ WgT, const __bf16* __restrict__ WuT,
                     const int* __restrict__ idx, const int* __restrict__ counts,
                     const int* __restrict__ offs, __bf16* __restrict__ H, int N) {
    const int e   = blockIdx.z;
    const int cnt = counts[e];
    const int row_start = blockIdx.x * 128;
    if (row_start >= cnt) return;
    const int n0    = blockIdx.y * 64;
    const int off_e = offs[e];

    __shared__ int sTok[128];
    __shared__ __align__(16) __bf16 As[2][128 * 32];
    __shared__ __align__(16) __bf16 Bg[2][64 * 32];
    __shared__ __align__(16) __bf16 Bu[2][64 * 32];

    const int tid  = threadIdx.x;
    const int lane = tid & 31;
    const int wv   = tid >> 5;
    const int wm   = wv & 3;
    const int wn   = wv >> 2;

    if (tid < 128) {
        const int pos = row_start + tid;
        sTok[tid] = idx[e * N + min(pos, cnt - 1)];
    }
    __syncthreads();

    const __bf16* wgp = WgT + (size_t)e * DFF * DMODEL;   // [DFF][DMODEL]
    const __bf16* wup = WuT + (size_t)e * DFF * DMODEL;

    // stage K-tile k0 into buffer b: pure async byte copies
    auto stage = [&](int b, int k0) {
#pragma unroll
        for (int j = 0; j < 2; ++j) {                      // A: 128x32 bf16
            const int lin = tid * 2 + j;                   // 0..511 chunks
            const int row = lin >> 2, c = lin & 3;
            asyncCp16(ldsU32(&As[b][row * 32 + c * 8]),
                      Xb + (size_t)sTok[row] * DMODEL + k0 + c * 8);
        }
        {                                                  // Bg/Bu: 64x32 bf16
            const int nrow = tid >> 2, c = tid & 3;
            asyncCp16(ldsU32(&Bg[b][nrow * 32 + c * 8]),
                      wgp + (size_t)(n0 + nrow) * DMODEL + k0 + c * 8);
            asyncCp16(ldsU32(&Bu[b][nrow * 32 + c * 8]),
                      wup + (size_t)(n0 + nrow) * DMODEL + k0 + c * 8);
        }
        __builtin_prefetch(wgp + (size_t)(n0 + (tid >> 2)) * DMODEL + k0 + 64, 0, 1);
    };

    const v8f zacc = {};
    v8f cg[2][2], cu[2][2];
#pragma unroll
    for (int i = 0; i < 2; ++i)
#pragma unroll
        for (int j = 0; j < 2; ++j) { cg[i][j] = zacc; cu[i][j] = zacc; }

    stage(0, 0);
    waitAsync0();
    __syncthreads();

    const int KT = DMODEL / 32;
    for (int kt = 0; kt < KT; ++kt) {
        const int buf = kt & 1;
        if (kt + 1 < KT) stage(buf ^ 1, (kt + 1) * 32);

        v16bf a[2], bg[2], bu[2];
#pragma unroll
        for (int mf = 0; mf < 2; ++mf) a[mf] = ldsFrag(As[buf], wm * 32 + mf * 16, lane);
#pragma unroll
        for (int nf = 0; nf < 2; ++nf) {
            bg[nf] = ldsFrag(Bg[buf], wn * 32 + nf * 16, lane);
            bu[nf] = ldsFrag(Bu[buf], wn * 32 + nf * 16, lane);
        }
#pragma unroll
        for (int mf = 0; mf < 2; ++mf)
#pragma unroll
            for (int nf = 0; nf < 2; ++nf) {
                cg[mf][nf] = __builtin_amdgcn_wmma_f32_16x16x32_bf16(
                    false, a[mf], false, bg[nf], (short)0, cg[mf][nf], false, false);
                cu[mf][nf] = __builtin_amdgcn_wmma_f32_16x16x32_bf16(
                    false, a[mf], false, bu[nf], (short)0, cu[mf][nf], false, false);
            }
        waitAsync0();
        __syncthreads();
    }

    // epilogue: H = gelu_exact(G) * U (bf16, compact slot layout)
    const int half = lane >> 4, col = lane & 15;
#pragma unroll
    for (int mf = 0; mf < 2; ++mf)
#pragma unroll
        for (int nf = 0; nf < 2; ++nf) {
#pragma unroll
            for (int r = 0; r < 8; ++r) {
                const int m   = wm * 32 + mf * 16 + half * 8 + r;
                const int pos = row_start + m;
                if (pos < cnt) {
                    const float gv = cg[mf][nf][r];
                    const float uv = cu[mf][nf][r];
                    const float h  = 0.5f * gv * (1.f + erff(gv * 0.70710678118654752f)) * uv;
                    H[(size_t)(off_e + pos) * DFF + n0 + wn * 32 + nf * 16 + col] = (__bf16)h;
                }
            }
        }
}

// ------------------------------------------- GEMM2: down-proj + scatter ----
__global__ __launch_bounds__(256)
void moe_gemm_down(const __bf16* __restrict__ H, const __bf16* __restrict__ WdT,
                   const int* __restrict__ idx, const float* __restrict__ wgt,
                   const int* __restrict__ counts, const int* __restrict__ offs,
                   float* __restrict__ out, int N) {
    const int e   = blockIdx.z;
    const int cnt = counts[e];
    const int row_start = blockIdx.x * 128;
    if (row_start >= cnt) return;
    const int n0    = blockIdx.y * 64;
    const int off_e = offs[e];

    __shared__ int   sTok[128];
    __shared__ float sW[128];
    __shared__ __align__(16) __bf16 As[2][128 * 32];
    __shared__ __align__(16) __bf16 Bd[2][64 * 32];

    const int tid  = threadIdx.x;
    const int lane = tid & 31;
    const int wv   = tid >> 5;
    const int wm   = wv & 3;
    const int wn   = wv >> 2;

    if (tid < 128) {
        const int pos = row_start + tid;
        const int cp  = min(pos, cnt - 1);
        sTok[tid] = idx[e * N + cp];
        sW[tid]   = wgt[e * N + cp];
    }
    __syncthreads();

    const __bf16* wdp = WdT + (size_t)e * DMODEL * DFF;   // [DMODEL][DFF]
    const __bf16* hrow = H + (size_t)(off_e + row_start) * DFF;

    auto stage = [&](int b, int k0) {
#pragma unroll
        for (int j = 0; j < 2; ++j) {                      // A: H rows (bf16)
            const int lin = tid * 2 + j;
            const int row = lin >> 2, c = lin & 3;
            asyncCp16(ldsU32(&As[b][row * 32 + c * 8]),
                      hrow + (size_t)row * DFF + k0 + c * 8);
        }
        {                                                  // B: WdT rows
            const int nrow = tid >> 2, c = tid & 3;
            asyncCp16(ldsU32(&Bd[b][nrow * 32 + c * 8]),
                      wdp + (size_t)(n0 + nrow) * DFF + k0 + c * 8);
        }
        __builtin_prefetch(wdp + (size_t)(n0 + (tid >> 2)) * DFF + k0 + 64, 0, 1);
    };

    const v8f zacc = {};
    v8f c[2][2];
#pragma unroll
    for (int i = 0; i < 2; ++i)
#pragma unroll
        for (int j = 0; j < 2; ++j) c[i][j] = zacc;

    stage(0, 0);
    waitAsync0();
    __syncthreads();

    const int KT = DFF / 32;
    for (int kt = 0; kt < KT; ++kt) {
        const int buf = kt & 1;
        if (kt + 1 < KT) stage(buf ^ 1, (kt + 1) * 32);

        v16bf a[2], b[2];
#pragma unroll
        for (int mf = 0; mf < 2; ++mf) a[mf] = ldsFrag(As[buf], wm * 32 + mf * 16, lane);
#pragma unroll
        for (int nf = 0; nf < 2; ++nf) b[nf] = ldsFrag(Bd[buf], wn * 32 + nf * 16, lane);
#pragma unroll
        for (int mf = 0; mf < 2; ++mf)
#pragma unroll
            for (int nf = 0; nf < 2; ++nf)
                c[mf][nf] = __builtin_amdgcn_wmma_f32_16x16x32_bf16(
                    false, a[mf], false, b[nf], (short)0, c[mf][nf], false, false);
        waitAsync0();
        __syncthreads();
    }

    const int half = lane >> 4, col = lane & 15;
#pragma unroll
    for (int mf = 0; mf < 2; ++mf)
#pragma unroll
        for (int nf = 0; nf < 2; ++nf) {
#pragma unroll
            for (int r = 0; r < 8; ++r) {
                const int m   = wm * 32 + mf * 16 + half * 8 + r;
                const int pos = row_start + m;
                if (pos < cnt) {
                    const float val = c[mf][nf][r] * sW[m];
                    unsafeAtomicAdd(&out[(size_t)sTok[m] * DMODEL + n0 + wn * 32 + nf * 16 + col], val);
                }
            }
        }
}

// ---------------------------------------------------------------- launch ----
extern "C" void kernel_launch(void* const* d_in, const int* in_sizes, int n_in,
                              void* d_out, int out_size, void* d_ws, size_t ws_size,
                              hipStream_t stream) {
    const float* x    = (const float*)d_in[0];
    const float* gw   = (const float*)d_in[1];
    const float* Wg   = (const float*)d_in[2];
    const float* Wu   = (const float*)d_in[3];
    const float* Wd   = (const float*)d_in[4];
    float* out = (float*)d_out;

    const int N = in_sizes[0] / DMODEL;   // 8192 tokens

    // workspace carve-up
    char* p = (char*)d_ws;
    int*   counts = (int*)p;               p += 256;
    int*   offs   = (int*)p;               p += 256;
    int*   idx    = (int*)p;               p += sizeof(int)   * NE * (size_t)N;
    float* wgt    = (float*)p;             p += sizeof(float) * NE * (size_t)N;
    p = (char*)(((uintptr_t)p + 255) & ~(uintptr_t)255);
    __bf16* Xb    = (__bf16*)p;            p += sizeof(__bf16) * (size_t)N * DMODEL;
    __bf16* WgT   = (__bf16*)p;            p += sizeof(__bf16) * (size_t)NE * DFF * DMODEL;
    __bf16* WuT   = (__bf16*)p;            p += sizeof(__bf16) * (size_t)NE * DFF * DMODEL;
    __bf16* WdT   = (__bf16*)p;            p += sizeof(__bf16) * (size_t)NE * DMODEL * DFF;
    __bf16* H     = (__bf16*)p;            // (TOPK*N + 128) x DFF bf16

    hipMemsetAsync(counts, 0, 256, stream);
    hipMemsetAsync(out, 0, sizeof(float) * (size_t)out_size, stream);

    // one-time bf16 repack (X linear; weights transposed to [n][k] panels)
    pack_x<<<(N * DMODEL) / 2048, 256, 0, stream>>>(x, Xb);
    pack_transpose<<<dim3(DMODEL / 64, DFF / 64, NE), 256, 0, stream>>>(Wg, WgT, DMODEL, DFF);
    pack_transpose<<<dim3(DMODEL / 64, DFF / 64, NE), 256, 0, stream>>>(Wu, WuT, DMODEL, DFF);
    pack_transpose<<<dim3(DFF / 64, DMODEL / 64, NE), 256, 0, stream>>>(Wd, WdT, DFF, DMODEL);

    moe_router<<<N / 8, 256, 0, stream>>>(x, gw, counts, idx, wgt, N);
    moe_prefix<<<1, 32, 0, stream>>>(counts, offs);
    moe_gemm_gateup<<<dim3(N / 128, DFF / 64, NE), 256, 0, stream>>>(
        Xb, WgT, WuT, idx, counts, offs, H, N);
    moe_gemm_down<<<dim3(N / 128, DMODEL / 64, NE), 256, 0, stream>>>(
        H, WdT, idx, wgt, counts, offs, out, N);
}
